// DiffS6Mamba2_11209864642877
// MI455X (gfx1250) — compile-verified, matrix operand order checked by reading, not run on hardware
//
#include <hip/hip_runtime.h>
#include <hip/hip_bf16.h>
#include <math.h>

typedef __attribute__((ext_vector_type(16))) _Float16 v16h;
typedef __attribute__((ext_vector_type(8)))  _Float16 v8h;
typedef __attribute__((ext_vector_type(8)))  float    v8f;
typedef int v4i __attribute__((vector_size(16)));   // matches async-LDS builtin param

namespace {
constexpr int kDModel  = 1024;
constexpr int kDInner  = 2048;
constexpr int kDState  = 64;
constexpr int kHeadDim = 64;
constexpr int kNHeads  = 32;
constexpr int kChunk   = 256;
constexpr int kDInProj = 4256;   // 2*2048 + 2*64 + 32
constexpr int kConvDim = 2176;   // 2048 + 128
constexpr int kBatch   = 2;
constexpr int kSeqLen  = 2048;
constexpr int kRows    = kBatch * kSeqLen;   // 4096
constexpr int kNChunks = kSeqLen / kChunk;   // 8
constexpr int kNBC     = kBatch * kNChunks;  // 16
} // namespace

// ---------------------------------------------------------------------------
// Generic batched GEMM:  C[z][M,N] = A[z][M,K] * Bt[z][N,K]^T
// f32 inputs converted to f16 while staging into LDS (branch-free, masked);
// v_wmma_f32_16x16x32_f16 with f32 accumulation.
// Block = 256 threads = 8 waves; tile 128x128x32; wave grid 4(M) x 2(N).
// ---------------------------------------------------------------------------
__global__ __launch_bounds__(256)
void gemm_wmma(const float* __restrict__ A, int lda, long long sA,
               const float* __restrict__ B, int ldb, long long sB,
               float* __restrict__ C, int ldc, long long sC,
               int M, int N, int K)
{
    __shared__ __align__(16) _Float16 As[128 * 40];   // 128 rows x 32 K (+8 pad)
    __shared__ __align__(16) _Float16 Bs[128 * 40];

    A += (long long)blockIdx.z * sA;
    B += (long long)blockIdx.z * sB;
    C += (long long)blockIdx.z * sC;

    const int m0   = blockIdx.y * 128;
    const int n0   = blockIdx.x * 128;
    const int tid  = threadIdx.x;
    const int lane = tid & 31;
    const int w    = tid >> 5;
    const int wm   = (w & 3) * 32;   // wave row offset in tile
    const int wn   = (w >> 2) * 64;  // wave col offset in tile

    const v8f vzero = {0.f,0.f,0.f,0.f,0.f,0.f,0.f,0.f};
    v8f acc[2][4];
#pragma unroll
    for (int i = 0; i < 2; ++i)
#pragma unroll
        for (int j = 0; j < 4; ++j) acc[i][j] = vzero;

    const int lr   = tid >> 1;          // staging row 0..127
    const int lc   = (tid & 1) << 4;    // staging col 0 or 16
    const int frow = lane & 15;         // fragment row within 16
    const int kh   = (lane >> 4) << 3;  // fragment K sub-offset 0 or 8

    // Branch-free row guards: clamp pointer, scale by 0/1 mask.
    const int grA = m0 + lr;
    const float mA = (grA < M) ? 1.f : 0.f;
    const float* rowA = A + (long long)(grA < M ? grA : 0) * lda + lc;
    const int grB = n0 + lr;
    const float mB = (grB < N) ? 1.f : 0.f;
    const float* rowB = B + (long long)(grB < N ? grB : 0) * ldb + lc;
    _Float16* dA = &As[lr * 40 + lc];
    _Float16* dB = &Bs[lr * 40 + lc];

    for (int k0 = 0; k0 < K; k0 += 32) {
        // ---- stage A tile (f32 -> f16, masked) ----
        {
            __builtin_prefetch(rowA + k0 + 32, 0, 1);   // global_prefetch (speculative)
            const float4 v0 = *(const float4*)(rowA + k0);
            const float4 v1 = *(const float4*)(rowA + k0 + 4);
            const float4 v2 = *(const float4*)(rowA + k0 + 8);
            const float4 v3 = *(const float4*)(rowA + k0 + 12);
            v8h h0, h1;
            h0[0]=(_Float16)(v0.x*mA); h0[1]=(_Float16)(v0.y*mA);
            h0[2]=(_Float16)(v0.z*mA); h0[3]=(_Float16)(v0.w*mA);
            h0[4]=(_Float16)(v1.x*mA); h0[5]=(_Float16)(v1.y*mA);
            h0[6]=(_Float16)(v1.z*mA); h0[7]=(_Float16)(v1.w*mA);
            h1[0]=(_Float16)(v2.x*mA); h1[1]=(_Float16)(v2.y*mA);
            h1[2]=(_Float16)(v2.z*mA); h1[3]=(_Float16)(v2.w*mA);
            h1[4]=(_Float16)(v3.x*mA); h1[5]=(_Float16)(v3.y*mA);
            h1[6]=(_Float16)(v3.z*mA); h1[7]=(_Float16)(v3.w*mA);
            *(v8h*)dA = h0; *(v8h*)(dA + 8) = h1;
        }
        // ---- stage B tile (f32 -> f16, masked) ----
        {
            __builtin_prefetch(rowB + k0 + 32, 0, 1);
            const float4 v0 = *(const float4*)(rowB + k0);
            const float4 v1 = *(const float4*)(rowB + k0 + 4);
            const float4 v2 = *(const float4*)(rowB + k0 + 8);
            const float4 v3 = *(const float4*)(rowB + k0 + 12);
            v8h h0, h1;
            h0[0]=(_Float16)(v0.x*mB); h0[1]=(_Float16)(v0.y*mB);
            h0[2]=(_Float16)(v0.z*mB); h0[3]=(_Float16)(v0.w*mB);
            h0[4]=(_Float16)(v1.x*mB); h0[5]=(_Float16)(v1.y*mB);
            h0[6]=(_Float16)(v1.z*mB); h0[7]=(_Float16)(v1.w*mB);
            h1[0]=(_Float16)(v2.x*mB); h1[1]=(_Float16)(v2.y*mB);
            h1[2]=(_Float16)(v2.z*mB); h1[3]=(_Float16)(v2.w*mB);
            h1[4]=(_Float16)(v3.x*mB); h1[5]=(_Float16)(v3.y*mB);
            h1[6]=(_Float16)(v3.z*mB); h1[7]=(_Float16)(v3.w*mB);
            *(v8h*)dB = h0; *(v8h*)(dB + 8) = h1;
        }
        __syncthreads();

        // ---- fragments: A frag lane layout == B frag lane layout (Bt trick)
        v16h af[2], bf[4];
#pragma unroll
        for (int mt = 0; mt < 2; ++mt) {
            const _Float16* p = &As[(wm + mt * 16 + frow) * 40 + kh];
            ((v8h*)&af[mt])[0] = *(const v8h*)p;
            ((v8h*)&af[mt])[1] = *(const v8h*)(p + 16);
        }
#pragma unroll
        for (int nt = 0; nt < 4; ++nt) {
            const _Float16* p = &Bs[(wn + nt * 16 + frow) * 40 + kh];
            ((v8h*)&bf[nt])[0] = *(const v8h*)p;
            ((v8h*)&bf[nt])[1] = *(const v8h*)(p + 16);
        }
#pragma unroll
        for (int mt = 0; mt < 2; ++mt)
#pragma unroll
            for (int nt = 0; nt < 4; ++nt)
                acc[mt][nt] = __builtin_amdgcn_wmma_f32_16x16x32_f16(
                    false, af[mt], false, bf[nt], (short)0, acc[mt][nt], false, false);
        __syncthreads();
    }

    // ---- store C (v8f layout: lanes 0-15 -> M=r, lanes 16-31 -> M=8+r) ----
#pragma unroll
    for (int mt = 0; mt < 2; ++mt) {
        const int r0 = m0 + wm + mt * 16 + ((lane >> 4) << 3);
#pragma unroll
        for (int nt = 0; nt < 4; ++nt) {
            const int col = n0 + wn + nt * 16 + (lane & 15);
            if (col < N) {
#pragma unroll
                for (int r = 0; r < 8; ++r) {
                    const int row = r0 + r;
                    if (row < M) C[(long long)row * ldc + col] = acc[mt][nt][r];
                }
            }
        }
    }
}

// ---------------------------------------------------------------------------
// dt = softplus(zx[:, 4224+h] + dt_bias[h])
// ---------------------------------------------------------------------------
__global__ __launch_bounds__(256)
void dt_softplus(const float* __restrict__ zx, const float* __restrict__ dt_bias,
                 float* __restrict__ dt)
{
    const int i = blockIdx.x * 256 + threadIdx.x;          // over 4096*32
    const int h = i & 31, row = i >> 5;
    const float x = zx[(long long)row * kDInProj + (kDInner + kConvDim) + h] + dt_bias[h];
    dt[i] = (x > 20.f) ? x : log1pf(__expf(x));
}

// ---------------------------------------------------------------------------
// Depthwise causal conv (K=4) + SiLU over xBC slice of zxbcdt.
// ---------------------------------------------------------------------------
__global__ __launch_bounds__(256)
void conv_silu(const float* __restrict__ zx, const float* __restrict__ w,
               const float* __restrict__ b, float* __restrict__ out)
{
    const int i = blockIdx.x * 256 + threadIdx.x;          // over 4096*2176
    const int c = i % kConvDim;
    const int row = i / kConvDim;
    const int l = row & (kSeqLen - 1);
    float acc = b[c];
#pragma unroll
    for (int j = 0; j < 4; ++j) {
        const int ls = l - 3 + j;
        if (ls >= 0)
            acc += w[c * 4 + j] * zx[(long long)(row - l + ls) * kDInProj + kDInner + c];
    }
    out[i] = acc / (1.f + __expf(-acc));
}

// ---------------------------------------------------------------------------
// dAcs[b,c,h,q] = cumsum_q( dt[b, c*256+q, h] * (-exp(A_log[h])) )
// ---------------------------------------------------------------------------
__global__ __launch_bounds__(256)
void chunk_cumsum(const float* __restrict__ dt, const float* __restrict__ A_log,
                  float* __restrict__ dAcs)
{
    const int bid = blockIdx.x;       // (b*8 + c)*32 + h
    const int h = bid & 31, bc = bid >> 5;
    const int q = threadIdx.x;
    __shared__ float s[256];
    const float Ah = -__expf(A_log[h]);
    s[q] = dt[(bc * 256 + q) * kNHeads + h] * Ah;
    __syncthreads();
    for (int off = 1; off < 256; off <<= 1) {
        const float t = (q >= off) ? s[q - off] : 0.f;
        __syncthreads();
        s[q] += t;
        __syncthreads();
    }
    dAcs[bid * 256 + q] = s[q];
}

// ---------------------------------------------------------------------------
// states[b,c,h,p,n] = sum_k exp(dAcs[255]-dAcs[k]) * dt[k] * x[k,p] * B[k,n]
// ---------------------------------------------------------------------------
__global__ __launch_bounds__(256)
void chunk_states(const float* __restrict__ conv, const float* __restrict__ dt,
                  const float* __restrict__ dAcs, float* __restrict__ states)
{
    const int bid = blockIdx.x;       // (b*8 + c)*32 + h
    const int h = bid & 31, bc = bid >> 5;
    const int t = threadIdx.x;
    __shared__ float coef[256];
    {
        const float dlast = dAcs[bid * 256 + 255];
        coef[t] = __expf(dlast - dAcs[bid * 256 + t]) * dt[(bc * 256 + t) * kNHeads + h];
    }
    __syncthreads();
    const int p = t >> 2;
    const int n0 = (t & 3) * 16;
    float acc[16];
#pragma unroll
    for (int j = 0; j < 16; ++j) acc[j] = 0.f;
    for (int k = 0; k < 256; ++k) {
        const long long row = (long long)(bc * 256 + k) * kConvDim;
        const float xv = conv[row + h * kHeadDim + p] * coef[k];
        const float* Bp = &conv[row + kDInner + n0];
#pragma unroll
        for (int j = 0; j < 16; ++j) acc[j] += xv * Bp[j];
    }
    float* out = &states[(long long)bid * 4096 + p * 64 + n0];
#pragma unroll
    for (int j = 0; j < 16; ++j) out[j] = acc[j];
}

// ---------------------------------------------------------------------------
// Sequential recurrence over chunks:  prev[c] = S;  S = decay_c*S + states[c]
// ---------------------------------------------------------------------------
__global__ __launch_bounds__(256)
void chunk_scan_states(const float* __restrict__ states, const float* __restrict__ dAcs,
                       float* __restrict__ prev)
{
    const int i = blockIdx.x * 256 + threadIdx.x;  // over 2*32*64*64
    const int pn = i & 4095;
    const int rest = i >> 12;                      // b*32 + h
    const int h = rest & 31, b = rest >> 5;
    float S = 0.f;
    for (int c = 0; c < kNChunks; ++c) {
        const int bid = (b * kNChunks + c) * kNHeads + h;
        prev[(long long)bid * 4096 + pn] = S;
        const float decay = __expf(dAcs[bid * 256 + 255]);
        S = decay * S + states[(long long)bid * 4096 + pn];
    }
}

// ---------------------------------------------------------------------------
// Intra-chunk: Y[q,p] = sum_k M[q,k] * X[k,p],
//   M[q,k] = CB[q,k]*exp(dAcs[q]-dAcs[k])*dt[k]  (k<=q, else 0)
// M (256x256) and X^T (64x256) built in f16 LDS; computed with WMMA.
// Dynamic LDS = 256*264 + 64*264 halves + 512 floats = 171,008 B (< 320 KB).
// ---------------------------------------------------------------------------
__global__ __launch_bounds__(256)
void chunk_intra(const float* __restrict__ conv, const float* __restrict__ dt,
                 const float* __restrict__ dAcs, const float* __restrict__ CB,
                 float* __restrict__ y)
{
    extern __shared__ __align__(16) char smem[];
    _Float16* Msh = (_Float16*)smem;            // 256 x 264
    _Float16* Xt  = Msh + 256 * 264;            // 64  x 264   (X transposed)
    float* dAs = (float*)(Xt + 64 * 264);       // 256
    float* dts = dAs + 256;                     // 256

    const int bid = blockIdx.x;                 // (b*8 + c)*32 + h
    const int h = bid & 31, bc = bid >> 5;
    const int tid = threadIdx.x;

    dAs[tid] = dAcs[bid * 256 + tid];
    dts[tid] = dt[(bc * 256 + tid) * kNHeads + h];
    __syncthreads();

    {   // build masked-decay matrix row q = tid (branch-free, v8h stores)
        const int q = tid;
        const float dq = dAs[q];
        const float* cbrow = &CB[(long long)bc * 65536 + q * 256];
        _Float16* mrow = &Msh[q * 264];
        for (int k8 = 0; k8 < 256; k8 += 8) {
            v8h hv;
#pragma unroll
            for (int j = 0; j < 8; ++j) {
                const int k = k8 + j;
                const float v = (k <= q) ? cbrow[k] * __expf(dq - dAs[k]) * dts[k] : 0.f;
                hv[j] = (_Float16)v;
            }
            *(v8h*)(mrow + k8) = hv;
        }
    }
    {   // build X^T : row k = tid of X goes to column tid of Xt
        const float* xrow = &conv[(long long)(bc * 256 + tid) * kConvDim + h * kHeadDim];
#pragma unroll 8
        for (int p = 0; p < 64; ++p) Xt[p * 264 + tid] = (_Float16)xrow[p];
    }
    __syncthreads();

    const int lane = tid & 31, w = tid >> 5;
    const int frow = lane & 15;
    const int kh = (lane >> 4) << 3;
    const v8f vzero = {0.f,0.f,0.f,0.f,0.f,0.f,0.f,0.f};
    v8f acc[2][4];
#pragma unroll
    for (int i = 0; i < 2; ++i)
#pragma unroll
        for (int j = 0; j < 4; ++j) acc[i][j] = vzero;

    for (int k0 = 0; k0 < 256; k0 += 32) {
        v16h af[2], bf[4];
#pragma unroll
        for (int mt = 0; mt < 2; ++mt) {
            const _Float16* p = &Msh[(w * 32 + mt * 16 + frow) * 264 + k0 + kh];
            ((v8h*)&af[mt])[0] = *(const v8h*)p;
            ((v8h*)&af[mt])[1] = *(const v8h*)(p + 16);
        }
#pragma unroll
        for (int nt = 0; nt < 4; ++nt) {
            const _Float16* p = &Xt[(nt * 16 + frow) * 264 + k0 + kh];
            ((v8h*)&bf[nt])[0] = *(const v8h*)p;
            ((v8h*)&bf[nt])[1] = *(const v8h*)(p + 16);
        }
#pragma unroll
        for (int mt = 0; mt < 2; ++mt)
#pragma unroll
            for (int nt = 0; nt < 4; ++nt)
                acc[mt][nt] = __builtin_amdgcn_wmma_f32_16x16x32_f16(
                    false, af[mt], false, bf[nt], (short)0, acc[mt][nt], false, false);
    }

#pragma unroll
    for (int mt = 0; mt < 2; ++mt) {
        const int q0 = w * 32 + mt * 16 + ((lane >> 4) << 3);
#pragma unroll
        for (int nt = 0; nt < 4; ++nt) {
            const int p = nt * 16 + (lane & 15);
#pragma unroll
            for (int r = 0; r < 8; ++r) {
                const int q = q0 + r;
                y[(long long)(bc * 256 + q) * kDInner + h * kHeadDim + p] = acc[mt][nt][r];
            }
        }
    }
}

// ---------------------------------------------------------------------------
// Inter-chunk + D*x residual:
//   y[q,p] += exp(dAcs[q]) * sum_n C[q,n]*prev[p,n]  +  D_h * x[q,p]
// prev tile (16 KB) staged into LDS with GLOBAL_LOAD_ASYNC_TO_LDS_B128
// (ASYNCcnt path) when the builtin is available, else plain copy.
// ---------------------------------------------------------------------------
__global__ __launch_bounds__(256)
void chunk_inter(const float* __restrict__ conv, const float* __restrict__ dAcs,
                 const float* __restrict__ prev, const float* __restrict__ Dv,
                 float* __restrict__ y)
{
    __shared__ __align__(16) float prevSh[64 * 64];
    const int bid = blockIdx.x;                 // (b*8 + c)*32 + h
    const int h = bid & 31, bc = bid >> 5;
    const int tid = threadIdx.x;
    const float* gsrc = &prev[(long long)bid * 4096];
#if defined(__AMDGCN__) && __has_builtin(__builtin_amdgcn_global_load_async_to_lds_b128)
    {
        // 4096 floats = 1024 x 16B transfers; 4 per thread.
#pragma unroll
        for (int it = 0; it < 4; ++it) {
            const int i = (tid + it * 256) * 4;
            __builtin_amdgcn_global_load_async_to_lds_b128(
                (v4i*)(gsrc + i), (v4i*)(prevSh + i), 0, 0);
        }
#if __has_builtin(__builtin_amdgcn_s_wait_asynccnt)
        __builtin_amdgcn_s_wait_asynccnt(0);
#else
        asm volatile("s_wait_asynccnt 0" ::: "memory");
#endif
    }
#else
    for (int i = tid; i < 4096; i += 256) prevSh[i] = gsrc[i];
#endif
    __syncthreads();

    const int q = tid;
    float acc[64];
#pragma unroll
    for (int p = 0; p < 64; ++p) acc[p] = 0.f;
    const float* crow = &conv[(long long)(bc * 256 + q) * kConvDim + kDInner + kDState];
    for (int n = 0; n < 64; ++n) {
        const float cn = crow[n];
#pragma unroll
        for (int p = 0; p < 64; ++p) acc[p] += cn * prevSh[p * 64 + n];
    }
    const float e = __expf(dAcs[bid * 256 + q]);
    const float Dh = Dv[h];
    const float* xrow = &conv[(long long)(bc * 256 + q) * kConvDim + h * kHeadDim];
    float* yrow = &y[(long long)(bc * 256 + q) * kDInner + h * kHeadDim];
#pragma unroll 8
    for (int p = 0; p < 64; ++p) yrow[p] += e * acc[p] + Dh * xrow[p];
}

// ---------------------------------------------------------------------------
// Gated RMS norm:  g = y * silu(z);  y = g * rsqrt(mean(g^2)+eps) * norm_w
// ---------------------------------------------------------------------------
__global__ __launch_bounds__(256)
void gated_rmsnorm_k(const float* __restrict__ zx, const float* __restrict__ norm_w,
                     float* __restrict__ y)
{
    const int row = blockIdx.x;
    const int tid = threadIdx.x;
    __shared__ float red[256];
    const float* zrow = &zx[(long long)row * kDInProj];
    float* yrow = &y[(long long)row * kDInner];
    float g[8]; float ss = 0.f;
#pragma unroll
    for (int j = 0; j < 8; ++j) {
        const int f = tid + j * 256;
        const float z = zrow[f];
        const float gv = yrow[f] * (z / (1.f + __expf(-z)));
        g[j] = gv; ss += gv * gv;
    }
    red[tid] = ss; __syncthreads();
    for (int off = 128; off > 0; off >>= 1) {
        if (tid < off) red[tid] += red[tid + off];
        __syncthreads();
    }
    const float scale = rsqrtf(red[0] / (float)kDInner + 1e-5f);
#pragma unroll
    for (int j = 0; j < 8; ++j) {
        const int f = tid + j * 256;
        yrow[f] = g[j] * scale * norm_w[f];
    }
}

// ---------------------------------------------------------------------------
// attn = rmsnorm(y1 - lam*y2) * subln_w * (1-lambda_init)  -> stored in y1
// ---------------------------------------------------------------------------
__global__ __launch_bounds__(256)
void combine_rms(float* __restrict__ y1, const float* __restrict__ y2,
                 const float* __restrict__ subln_w, const float* __restrict__ lambda_q1)
{
    const int row = blockIdx.x;
    const int tid = threadIdx.x;
    __shared__ float red[256];
    const float lamInit = 0.8f - 0.6f * __expf(-0.3f);
    const float lam = 1.f / (1.f + __expf(-lambda_q1[0])) + lamInit;
    float a[8]; float ss = 0.f;
#pragma unroll
    for (int j = 0; j < 8; ++j) {
        const int f = tid + j * 256;
        const float v = y1[(long long)row * kDInner + f] - lam * y2[(long long)row * kDInner + f];
        a[j] = v; ss += v * v;
    }
    red[tid] = ss; __syncthreads();
    for (int off = 128; off > 0; off >>= 1) {
        if (tid < off) red[tid] += red[tid + off];
        __syncthreads();
    }
    const float scale = rsqrtf(red[0] / (float)kDInner + 1e-5f) * (1.f - lamInit);
#pragma unroll
    for (int j = 0; j < 8; ++j) {
        const int f = tid + j * 256;
        y1[(long long)row * kDInner + f] = a[j] * scale * subln_w[f];
    }
}

// ---------------------------------------------------------------------------
extern "C" void kernel_launch(void* const* d_in, const int* in_sizes, int n_in,
                              void* d_out, int out_size, void* d_ws, size_t ws_size,
                              hipStream_t stream) {
    (void)in_sizes; (void)n_in; (void)out_size; (void)ws_size;
    const float* u        = (const float*)d_in[0];
    const float* W_in1    = (const float*)d_in[1];
    const float* W_in2    = (const float*)d_in[2];
    const float* conv_w   = (const float*)d_in[3];
    const float* conv_b   = (const float*)d_in[4];
    const float* dt1_bias = (const float*)d_in[5];
    const float* dt2_bias = (const float*)d_in[6];
    const float* A1_log   = (const float*)d_in[7];
    const float* A2_log   = (const float*)d_in[8];
    const float* D1       = (const float*)d_in[9];
    const float* D2       = (const float*)d_in[10];
    const float* norm_w   = (const float*)d_in[11];
    const float* subln_w  = (const float*)d_in[12];
    const float* lambda_q1= (const float*)d_in[13];
    const float* W_out    = (const float*)d_in[14];
    float* out = (float*)d_out;

    // Workspace layout (floats)
    float* ws = (float*)d_ws;
    size_t o = 0;
    float* zx     = ws + o; o += (size_t)kRows * kDInProj;            // 17,432,576
    float* conv   = ws + o; o += (size_t)kRows * kConvDim;            //  8,912,896
    float* dt     = ws + o; o += (size_t)kRows * kNHeads;             //    131,072
    float* dAcs   = ws + o; o += (size_t)kNBC * kNHeads * kChunk;     //    131,072
    float* CB     = ws + o; o += (size_t)kNBC * kChunk * kChunk;      //  1,048,576
    float* states = ws + o; o += (size_t)kNBC * kNHeads * 4096;       //  2,097,152
    float* prev   = ws + o; o += (size_t)kNBC * kNHeads * 4096;       //  2,097,152
    float* y1     = ws + o; o += (size_t)kRows * kDInner;             //  8,388,608
    float* y2     = ws + o; o += (size_t)kRows * kDInner;             //  8,388,608

    const int intraSmem = (256 * 264 + 64 * 264) * 2 + 512 * 4;       // 171,008 B

    auto branch = [&](const float* W_in, const float* dt_bias, const float* A_log,
                      const float* Dv, float* y) {
        // in-projection: zx = u @ W_in^T
        dim3 gIn((kDInProj + 127) / 128, kRows / 128, 1);
        gemm_wmma<<<gIn, 256, 0, stream>>>(u, kDModel, 0, W_in, kDModel, 0,
                                           zx, kDInProj, 0, kRows, kDInProj, kDModel);
        dt_softplus<<<(kRows * kNHeads) / 256, 256, 0, stream>>>(zx, dt_bias, dt);
        conv_silu<<<(kRows * kConvDim) / 256, 256, 0, stream>>>(zx, conv_w, conv_b, conv);
        chunk_cumsum<<<kNBC * kNHeads, 256, 0, stream>>>(dt, A_log, dAcs);
        // CB[bc][q,k] = C[q,:] . B[k,:]   (batched over bc)
        dim3 gCB(2, 2, kNBC);
        gemm_wmma<<<gCB, 256, 0, stream>>>(conv + kDInner + kDState, kConvDim,
                                           (long long)kChunk * kConvDim,
                                           conv + kDInner, kConvDim,
                                           (long long)kChunk * kConvDim,
                                           CB, kChunk, (long long)kChunk * kChunk,
                                           kChunk, kChunk, kDState);
        chunk_states<<<kNBC * kNHeads, 256, 0, stream>>>(conv, dt, dAcs, states);
        chunk_scan_states<<<(kBatch * kNHeads * 4096) / 256, 256, 0, stream>>>(states, dAcs, prev);
        chunk_intra<<<kNBC * kNHeads, 256, intraSmem, stream>>>(conv, dt, dAcs, CB, y);
        chunk_inter<<<kNBC * kNHeads, 256, 0, stream>>>(conv, dAcs, prev, Dv, y);
        gated_rmsnorm_k<<<kRows, 256, 0, stream>>>(zx, norm_w, y);
    };

    branch(W_in1, dt1_bias, A1_log, D1, y1);
    branch(W_in2, dt2_bias, A2_log, D2, y2);

    combine_rms<<<kRows, 256, 0, stream>>>(y1, y2, subln_w, lambda_q1);

    // out-projection: out = attn @ W_out^T
    dim3 gOut(kDModel / 128, kRows / 128, 1);
    gemm_wmma<<<gOut, 256, 0, stream>>>(y1, kDInner, 0, W_out, kDInner, 0,
                                        out, kDModel, 0, kRows, kDModel, kDInner);
}
